// CrossModalAttention_13151189860952
// MI455X (gfx1250) — matrix operand order repaired and checked by reading, not compiled, not run on hardware
//
#include <hip/hip_runtime.h>
#include <hip/hip_bf16.h>

typedef __attribute__((ext_vector_type(16))) _Float16 v16h;
typedef __attribute__((ext_vector_type(8)))  _Float16 v8h;
typedef __attribute__((ext_vector_type(8)))  float    v8f;
typedef __attribute__((ext_vector_type(4)))  float    v4f;

static constexpr int Bv = 8, Cv = 512, Hv = 48, Wv = 48, Nv = Hv * Wv; // N = 2304

#define DEVI __device__ __forceinline__

// ---------------------------------------------------------------------------
// WMMA f16 16x16x32 fragment helpers (wave32 layouts per CDNA5 ISA 7.12.2)
// ---------------------------------------------------------------------------

// A-matrix 16x32 (MxK), element (m,k); memory layout row-major: p[m*ld + k].
// Lane half h=lane/16, m=lane%16: elements 0..7 -> k = h*8 + 0..7,
// elements 8..15 -> k = 16 + h*8 + 0..7  (two contiguous 16B runs).
DEVI v16h ldA16(const _Float16* p, int ld, int lane) {
  const int hb = (lane >> 4) & 1, m = lane & 15;
  const _Float16* r = p + (size_t)m * ld + hb * 8;
  v8h lo = *(const v8h*)(r);
  v8h hi = *(const v8h*)(r + 16);
  return __builtin_shufflevector(lo, hi, 0,1,2,3,4,5,6,7,8,9,10,11,12,13,14,15);
}

// B-matrix 32x16 (KxN), element (k,n) stored k-contiguous: p[n*ld + k].
// Lane half h: elements 0..7 -> k = h*8 + 0..7, elements 8..15 -> k = 16 + h*8 + 0..7.
DEVI v16h ldBk16(const _Float16* p, int ld, int lane) {
  const int hb = (lane >> 4) & 1, n = lane & 15;
  const _Float16* r = p + (size_t)n * ld + hb * 8;
  v8h lo = *(const v8h*)(r);
  v8h hi = *(const v8h*)(r + 16);
  return __builtin_shufflevector(lo, hi, 0,1,2,3,4,5,6,7,8,9,10,11,12,13,14,15);
}

DEVI v8f wmma16(v16h a, v16h b, v8f c) {
  return __builtin_amdgcn_wmma_f32_16x16x32_f16(false, a, false, b, (short)0, c,
                                                false, false);
}

// ---------------------------------------------------------------------------
// Projection GEMM:  dst[b][m][n] (or transposed) = sum_k w[m][k]*src[b][k][n]+bias[m]
// M = C = 512, K = 512, N = 2304.  Block = 256 threads = 8 waves, 64x64 tile.
// Double-buffered LDS: one barrier per k-step, global loads for tile i+1
// overlap the WMMAs of tile i.  SRCF16 templated: no runtime branch, no
// f16->f32->f16 round trip for the K/V projections.
// ---------------------------------------------------------------------------
template <int SRCF16>
__global__ __launch_bounds__(256) void proj_kernel(
    const float* __restrict__ w, const float* __restrict__ bias,
    const void* __restrict__ src,
    _Float16* __restrict__ dst, int transposeOut) {
  constexpr int LDA = 40;  // 80B row stride: 16B aligned, staggers LDS banks
  constexpr int LDB = 40;
  __shared__ __align__(16) _Float16 As[2][64 * LDA];   // [m][k]
  __shared__ __align__(16) _Float16 Bt[2][64 * LDB];   // [n][k] (transposed)

  const int tid = threadIdx.x, lane = tid & 31, wid = tid >> 5;
  const int b = blockIdx.z;
  const int m0 = blockIdx.y * 64, n0 = blockIdx.x * 64;
  const int msub = wid & 3, nsub = wid >> 2;   // 4 (m) x 2 (n) wave grid
  const float*    srcF = (const float*)src;
  const _Float16* srcH = (const _Float16*)src;
  const size_t srcBase = (size_t)b * Cv * Nv;

  // Per-thread staging assignments (fixed across the k-loop)
  const int ar = tid >> 2, acb = (tid & 3) * 8;        // A: 8 contiguous k's of one row
  const int bc = tid & 63, brb = (tid >> 6) * 8;       // B: 8-row column segment

  v4f ax0, ax1;          // staged A values (f32)
  _Float16 bh[8];        // staged B values (f16)

  auto loadRegs = [&](int k0) {
    const float* pw = w + (size_t)(m0 + ar) * Cv + k0 + acb;
    ax0 = *(const v4f*)pw;
    ax1 = *(const v4f*)(pw + 4);
    if (k0 + 32 < Cv) __builtin_prefetch(pw + 32, 0, 1);
    const size_t g0 = srcBase + (size_t)(k0 + brb) * Nv + n0 + bc;
    if (SRCF16) {
#pragma unroll
      for (int i = 0; i < 8; ++i) bh[i] = srcH[g0 + (size_t)i * Nv];
    } else {
#pragma unroll
      for (int i = 0; i < 8; ++i) bh[i] = (_Float16)srcF[g0 + (size_t)i * Nv];
    }
  };
  auto storeBuf = [&](int bufi) {
    v8h av;
#pragma unroll
    for (int i = 0; i < 4; ++i) { av[i] = (_Float16)ax0[i]; av[4 + i] = (_Float16)ax1[i]; }
    *(v8h*)&As[bufi][ar * LDA + acb] = av;
    v8h bvv;
#pragma unroll
    for (int i = 0; i < 8; ++i) bvv[i] = bh[i];
    *(v8h*)&Bt[bufi][bc * LDB + brb] = bvv;
  };

  loadRegs(0);
  storeBuf(0);
  __syncthreads();

  v8f acc0 = {}, acc1 = {};
  constexpr int NK = Cv / 32;
  for (int it = 0; it < NK; ++it) {
    const int cur = it & 1;
    const bool more = (it + 1) < NK;
    if (more) loadRegs((it + 1) * 32);   // issue next tile's global loads
    v16h a  = ldA16(&As[cur][(msub * 16) * LDA], LDA, lane);
    v16h b0 = ldBk16(&Bt[cur][(nsub * 32) * LDB], LDB, lane);
    v16h b1 = ldBk16(&Bt[cur][(nsub * 32 + 16) * LDB], LDB, lane);
    acc0 = wmma16(a, b0, acc0);          // overlaps with outstanding loads
    acc1 = wmma16(a, b1, acc1);
    if (more) storeBuf(1 - cur);
    __syncthreads();                     // single barrier per k-step
  }

  // Epilogue: C/D layout -> row m = e + 8*h (contiguous in memory when
  // transposed), col n = lane&15.
  const int hb = (lane >> 4) & 1, nl = lane & 15;
  const int mrow0 = m0 + msub * 16 + 8 * hb;
  v8f bb = *(const v8f*)(bias + mrow0);
  if (transposeOut) {
#pragma unroll
    for (int t = 0; t < 2; ++t) {
      int n = n0 + nsub * 32 + t * 16 + nl;
      v8f vv = (t ? acc1 : acc0) + bb;
      v8h hv;
#pragma unroll
      for (int e = 0; e < 8; ++e) hv[e] = (_Float16)vv[e];
      *(v8h*)(dst + ((size_t)b * Nv + n) * Cv + mrow0) = hv;   // one 16B store
    }
  } else {
#pragma unroll
    for (int t = 0; t < 2; ++t) {
      v8f vv = (t ? acc1 : acc0) + bb;
      int n = n0 + nsub * 32 + t * 16 + nl;
#pragma unroll
      for (int e = 0; e < 8; ++e)
        dst[((size_t)b * Cv + mrow0 + e) * (size_t)Nv + n] = (_Float16)vv[e];
    }
  }
}

// ---------------------------------------------------------------------------
// Scores + softmax:  S = scale * Qt x K  (one wave per 16-query tile).
// Writes unnormalized P = exp(S - rowmax) in place (f16) + 1/rowsum (f32).
// ---------------------------------------------------------------------------
__global__ __launch_bounds__(256) void scores_kernel(
    const _Float16* __restrict__ qt,   // [B][N][C]
    const _Float16* __restrict__ kt,   // [B][N][C]
    _Float16* __restrict__ sp,         // [B][N][N]  scores -> probs (in place)
    float* __restrict__ invsum,        // [B][N]
    const float* __restrict__ scale) {
  const int tid = threadIdx.x, lane = tid & 31, wid = tid >> 5;
  const int b = blockIdx.y;
  const int q0 = (blockIdx.x * 8 + wid) * 16;
  const float sc = scale[0];
  const _Float16* qbase = qt + ((size_t)b * Nv + q0) * Cv;
  const _Float16* kbase = kt + (size_t)b * Nv * Cv;
  _Float16* srow = sp + ((size_t)b * Nv + q0) * Nv;

  // Q tile: 16 queries x 512 channels = 16 A fragments held in registers.
  v16h qa[16];
#pragma unroll
  for (int kk = 0; kk < 16; ++kk) qa[kk] = ldA16(qbase + kk * 32, Cv, lane);

  const int hb = (lane >> 4) & 1, nl = lane & 15;
  float rmax[8];
#pragma unroll
  for (int e = 0; e < 8; ++e) rmax[e] = -3.0e38f;

  // Pass 1: raw scores, running row max.
  for (int j = 0; j < Nv / 16; ++j) {
    const _Float16* kb = kbase + (size_t)(j * 16) * Cv;
    __builtin_prefetch(kb + (size_t)16 * Cv + (lane << 6), 0, 1);
    v8f acc = {};
#pragma unroll
    for (int kk = 0; kk < 16; ++kk) {
      v16h bf = ldBk16(kb + kk * 32, Cv, lane);
      acc = wmma16(qa[kk], bf, acc);
    }
    acc = acc * sc;
#pragma unroll
    for (int e = 0; e < 8; ++e) {
      rmax[e] = fmaxf(rmax[e], acc[e]);
      srow[(size_t)(e + 8 * hb) * Nv + j * 16 + nl] = (_Float16)acc[e];
    }
  }
  // Row max across the 16 lanes that hold each row (wave32 halves).
#pragma unroll
  for (int m = 1; m < 16; m <<= 1) {
#pragma unroll
    for (int e = 0; e < 8; ++e)
      rmax[e] = fmaxf(rmax[e], __shfl_xor(rmax[e], m, 32));
  }

  // Pass 2: exponentiate in place (L2-hot). Load-all / exp-all / store-all so
  // the 8 strided loads issue before any wait.
  float rsum[8] = {0.f, 0.f, 0.f, 0.f, 0.f, 0.f, 0.f, 0.f};
  for (int j = 0; j < Nv / 16; ++j) {
    float sv[8];
#pragma unroll
    for (int e = 0; e < 8; ++e)
      sv[e] = (float)srow[(size_t)(e + 8 * hb) * Nv + j * 16 + nl];
    float pv[8];
#pragma unroll
    for (int e = 0; e < 8; ++e) { pv[e] = __expf(sv[e] - rmax[e]); rsum[e] += pv[e]; }
#pragma unroll
    for (int e = 0; e < 8; ++e)
      srow[(size_t)(e + 8 * hb) * Nv + j * 16 + nl] = (_Float16)pv[e];
  }
#pragma unroll
  for (int m = 1; m < 16; m <<= 1) {
#pragma unroll
    for (int e = 0; e < 8; ++e) rsum[e] += __shfl_xor(rsum[e], m, 32);
  }
  if (nl == 0) {
#pragma unroll
    for (int e = 0; e < 8; ++e)
      invsum[(size_t)b * Nv + q0 + e + 8 * hb] = 1.0f / rsum[e];
  }
}

// ---------------------------------------------------------------------------
// Output GEMM:  out[b][c][q] = img[b][c][q] + invsum[q] * sum_m P[q][m]*V[c][m]
// Direct-from-global fragments (everything is L2 resident).  Block tile
// 64q x 128c: each wave owns 16q x 64c so one A fragment feeds 4 WMMAs.
// ---------------------------------------------------------------------------
__global__ __launch_bounds__(256) void out_kernel(
    const _Float16* __restrict__ sp,   // [B][N][N]  probs (unnormalized)
    const _Float16* __restrict__ vmat, // [B][C][N]
    const float* __restrict__ invsum,  // [B][N]
    const float* __restrict__ img,     // [B][C][N]
    float* __restrict__ out) {         // [B][C][N]
  const int tid = threadIdx.x, lane = tid & 31, wid = tid >> 5;
  const int b = blockIdx.z;
  const int q0 = blockIdx.y * 64, c0 = blockIdx.x * 128;
  const int msub = wid & 3, nsub = wid >> 2;   // 4 (q) x 2 (c) wave grid

  const _Float16* abase = sp + ((size_t)b * Nv + q0 + msub * 16) * Nv;
  const _Float16* bbase = vmat + ((size_t)b * Cv + c0 + nsub * 64) * Nv;

  v8f acc[4] = {{}, {}, {}, {}};
  for (int k0 = 0; k0 < Nv; k0 += 32) {
    __builtin_prefetch(abase + k0 + 128, 0, 1);           // global_prefetch_b8
    __builtin_prefetch(bbase + k0 + 128, 0, 1);
    v16h a = ldA16(abase + k0, Nv, lane);
#pragma unroll
    for (int t = 0; t < 4; ++t) {
      v16h bf = ldBk16(bbase + (size_t)(t * 16) * Nv + k0, Nv, lane);
      acc[t] = wmma16(a, bf, acc[t]);
    }
  }

  // Rows of the C/D fragment are contiguous q positions in [B][C][N]: do
  // whole-fragment 32B vector loads/stores per lane.
  const int hb = (lane >> 4) & 1, nl = lane & 15;
  const int qrow0 = q0 + msub * 16 + 8 * hb;
  v8f invv = *(const v8f*)(invsum + (size_t)b * Nv + qrow0);
#pragma unroll
  for (int t = 0; t < 4; ++t) {
    int c = c0 + nsub * 64 + t * 16 + nl;
    size_t o = ((size_t)b * Cv + c) * (size_t)Nv + qrow0;
    v8f imgv = *(const v8f*)(img + o);
    v8f res = imgv + acc[t] * invv;
    *(v8f*)(out + o) = res;
  }
}

// ---------------------------------------------------------------------------
extern "C" void kernel_launch(void* const* d_in, const int* in_sizes, int n_in,
                              void* d_out, int out_size, void* d_ws, size_t ws_size,
                              hipStream_t stream) {
  (void)in_sizes; (void)n_in; (void)out_size; (void)ws_size;
  const float* img   = (const float*)d_in[0];
  const float* text  = (const float*)d_in[1];
  const float* w_tp  = (const float*)d_in[2];
  const float* b_tp  = (const float*)d_in[3];
  const float* w_q   = (const float*)d_in[4];
  const float* b_q   = (const float*)d_in[5];
  const float* w_k   = (const float*)d_in[6];
  const float* b_k   = (const float*)d_in[7];
  const float* w_v   = (const float*)d_in[8];
  const float* b_v   = (const float*)d_in[9];
  const float* scale = (const float*)d_in[10];
  float* out = (float*)d_out;

  char* ws = (char*)d_ws;
  const size_t S1 = (size_t)Bv * Cv * Nv * sizeof(_Float16);   // 18.9 MB each
  _Float16* tf = (_Float16*)(ws);                 // [B][C][N]
  _Float16* qt = (_Float16*)(ws + S1);            // [B][N][C]
  _Float16* kt = (_Float16*)(ws + 2 * S1);        // [B][N][C]
  _Float16* vv = (_Float16*)(ws + 3 * S1);        // [B][C][N]
  _Float16* sp = (_Float16*)(ws + 4 * S1);        // [B][N][N]  (84.9 MB)
  const size_t S2 = (size_t)Bv * Nv * Nv * sizeof(_Float16);
  float* invsum = (float*)(ws + 4 * S1 + S2);     // [B][N]

  dim3 blk(256);
  dim3 gProj(Nv / 64, Cv / 64, Bv);
  proj_kernel<0><<<gProj, blk, 0, stream>>>(w_tp, b_tp, text, tf, 0);
  proj_kernel<0><<<gProj, blk, 0, stream>>>(w_q,  b_q,  img,  qt, 1);
  proj_kernel<1><<<gProj, blk, 0, stream>>>(w_k,  b_k,  tf,   kt, 1);
  proj_kernel<1><<<gProj, blk, 0, stream>>>(w_v,  b_v,  tf,   vv, 0);

  dim3 gS(Nv / 16 / 8, Bv);
  scores_kernel<<<gS, blk, 0, stream>>>(qt, kt, sp, invsum, scale);

  dim3 gO(Cv / 128, Nv / 64, Bv);
  out_kernel<<<gO, blk, 0, stream>>>(sp, vv, invsum, img, out);
}
